// MTModel_64347200029299
// MI455X (gfx1250) — matrix operand order repaired and checked by reading, compiled, not run on hardware
//
#include <hip/hip_runtime.h>
#include <hip/hip_bf16.h>

// ---------------------------------------------------------------------------
// Fused BiLSTM-encoder + (degenerate-)attention decoder for gfx1250 (MI455X).
//
// Key algebraic fact: the reference applies softmax over a size-1 axis, so the
// attention weights are identically 1.0 and context = a.sum(axis=1), constant
// across decoder steps. Wa1/ba1/Wa2/ba2 are dead. We therefore only accumulate
// the sum of encoder hidden states (fwd & rev) and never store a[B,Tx,64].
//
// One wave32 owns 16 batch rows. All matmuls are v_wmma_f32_16x16x32_f16 with
// f32 accumulation; weights are pre-packed per-block into LDS in B-fragment
// order; h/s state is bounced C-layout -> A-layout through per-wave LDS.
// ---------------------------------------------------------------------------

typedef _Float16 f16;
typedef __attribute__((ext_vector_type(16))) _Float16 v16h;
typedef __attribute__((ext_vector_type(8)))  float    v8f;

#define TX   30
#define TY   10
#define NA   32
#define NS   64
#define VIN  37
#define VOUT 11
#define BATCH 16384

#define WAVES_PER_BLOCK 8
#define THREADS   (WAVES_PER_BLOCK * 32)
#define ROWS_PER_WAVE 16
#define ROWS_PER_BLOCK (WAVES_PER_BLOCK * ROWS_PER_WAVE)
#define NBLOCKS (BATCH / ROWS_PER_BLOCK)

#define FRAG_HALFS 512   // one 32x16 f16 B-fragment: 32 lanes x 16 halfs

// LDS plan (halfs unless noted):
//  BIHF 16 frags | BIHR 16 | BHHF 8 | BHHR 8 | WCIH 32 | WCHH 32 | WOB 2
//  WBUF 8 waves x (16x64) | biases 528 f32 | OBUF 8 waves x 256 f32
#define N_WEIGHT_FRAGS (16 + 16 + 8 + 8 + 32 + 32 + 2)
#define WBUF_HALFS (WAVES_PER_BLOCK * 16 * 64)
#define SMEM_BYTES ((N_WEIGHT_FRAGS * FRAG_HALFS + WBUF_HALFS) * 2 + 528 * 4 + WAVES_PER_BLOCK * 256 * 4)

// ---------------- fragment helpers ----------------

__device__ __forceinline__ v8f zero8() {
  v8f z;
#pragma unroll
  for (int j = 0; j < 8; ++j) z[j] = 0.0f;
  return z;
}
__device__ __forceinline__ v16h zero16h() {
  v16h z;
#pragma unroll
  for (int j = 0; j < 16; ++j) z[j] = (f16)0.0f;
  return z;
}
__device__ __forceinline__ v8f splat8(float b) {
  v8f z;
#pragma unroll
  for (int j = 0; j < 8; ++j) z[j] = b;
  return z;
}

__device__ __forceinline__ v8f wmma16(v16h a, v16h b, v8f c) {
  // D = A(16x32 f16) * B(32x16 f16) + C(16x16 f32)
  return __builtin_amdgcn_wmma_f32_16x16x32_f16(false, a, false, b, (short)0, c,
                                                false, false);
}

// B fragment pre-packed in LDS as [lane][16 halfs]
__device__ __forceinline__ v16h loadB(const f16* frag, int lane) {
  return *(const v16h*)(frag + lane * 16);
}

// A fragment (16x32, 16-bit) from row-major LDS buffer [16][64]:
// lane L: M = L&15; VGPR v<4 holds K = koff + 2v + 8*(L>>4) (+1),
//         VGPR v>=4 holds K = koff + 16 + 2(v-4) + 8*(L>>4) (+1).
__device__ __forceinline__ v16h loadA_lds(const f16* buf, int lane, int koff) {
  int m = lane & 15, hs = lane >> 4;
  const f16* r = buf + m * 64 + koff + 8 * hs;
  v16h a;
#pragma unroll
  for (int v = 0; v < 4; ++v) { a[2 * v] = r[2 * v]; a[2 * v + 1] = r[2 * v + 1]; }
#pragma unroll
  for (int v = 0; v < 4; ++v) { a[8 + 2 * v] = r[16 + 2 * v]; a[9 + 2 * v] = r[16 + 2 * v + 1]; }
  return a;
}

// A fragment for x[:,t,:] (f32 global, K padded 37->64, zero-filled)
__device__ __forceinline__ v16h loadA_x(const float* __restrict__ x, long rowbase,
                                        int t, int lane, int koff) {
  int m = lane & 15, hs = lane >> 4;
  const float* xr = x + ((rowbase + m) * TX + t) * VIN;
  v16h a;
#pragma unroll
  for (int v = 0; v < 8; ++v) {
    int kb = koff + 8 * hs + ((v < 4) ? 2 * v : (16 + 2 * (v - 4)));
    float lo = (kb     < VIN) ? xr[kb]     : 0.0f;
    float hi = (kb + 1 < VIN) ? xr[kb + 1] : 0.0f;
    a[2 * v]     = (f16)lo;
    a[2 * v + 1] = (f16)hi;
  }
  return a;
}

// C fragment (16x16 f32) -> f16 LDS buffer [16][64] at column offset noff.
// Element (M = j + 8*(lane>>4), N = noff + (lane&15)).
__device__ __forceinline__ void storeC_lds(f16* buf, int lane, int noff, v8f c) {
  int n = lane & 15, hs = lane >> 4;
#pragma unroll
  for (int j = 0; j < 8; ++j) buf[(j + 8 * hs) * 64 + noff + n] = (f16)c[j];
}

__device__ __forceinline__ float sigmoidf_(float v) {
  return 1.0f / (1.0f + __expf(-v));
}

// Pointwise LSTM update on matched C-layout fragments (PyTorch gate order).
__device__ __forceinline__ void lstm_pointwise(v8f gi, v8f gf, v8f gg, v8f go,
                                               v8f& h, v8f& c) {
#pragma unroll
  for (int j = 0; j < 8; ++j) {
    float i = sigmoidf_(gi[j]);
    float f = sigmoidf_(gf[j]);
    float g = tanhf(gg[j]);
    float o = sigmoidf_(go[j]);
    float cn = f * c[j] + i * g;
    c[j] = cn;
    h[j] = o * tanhf(cn);
  }
}

// Pack a row-major f32 weight W[N][Kstride] into f16 B-fragments:
// frag(kc, nt) lane L element s: n = nt*16 + (L&15), k = kc*32 + 8*(L>>4) + (s&7) + 16*(s>>3); value W[n][k] (0-padded).
__device__ void packB(const float* __restrict__ W, int Nvalid, int Kvalid,
                      int Kstride, int ntiles, int kchunks, f16* dst,
                      int tid, int nthreads) {
  int total = ntiles * kchunks * FRAG_HALFS;
  for (int i = tid; i < total; i += nthreads) {
    int fidx = i >> 9;
    int rem  = i & 511;
    int lane = rem >> 4;
    int nl   = rem & 15;
    int kc   = fidx / ntiles;
    int nt   = fidx - kc * ntiles;
    int kg = kc * 32 + 8 * (lane >> 4) + (nl & 7) + ((nl >> 3) << 4);
    int ng = nt * 16 + (lane & 15);
    float v = (kg < Kvalid && ng < Nvalid) ? W[(long)ng * Kstride + kg] : 0.0f;
    dst[i] = (f16)v;
  }
}

// ---------------- encoder scan (one direction) ----------------

__device__ void encoder_dir(const float* __restrict__ x, long rowbase, int lane,
                            bool rev, const f16* BIH, const f16* BHH,
                            const float* BIASG, f16* wbuf, v8f* sum) {
  v8f hc[2] = {zero8(), zero8()};
  v8f cc[2] = {zero8(), zero8()};
  v16h hA = zero16h();
  sum[0] = zero8();
  sum[1] = zero8();
#pragma unroll 1
  for (int t = 0; t < TX; ++t) {
    int tt = rev ? (TX - 1 - t) : t;
    if (t + 1 < TX) {
      int tn = rev ? (TX - 2 - t) : (t + 1);
      __builtin_prefetch(x + ((rowbase + (lane & 15)) * TX + tn) * VIN, 0, 0);
    }
    v16h xA0 = loadA_x(x, rowbase, tt, lane, 0);
    v16h xA1 = loadA_x(x, rowbase, tt, lane, 32);
    // gates[16x128]: tiles {i:0,1 f:2,3 g:4,5 o:6,7}; pass p covers h cols 16p..16p+15
#pragma unroll
    for (int p = 0; p < 2; ++p) {
      v8f acc[4];
#pragma unroll
      for (int q = 0; q < 4; ++q) {
        int tile = 2 * q + p;
        v8f c = splat8(BIASG[tile * 16 + (lane & 15)]);
        c = wmma16(xA0, loadB(BIH + (0 * 8 + tile) * FRAG_HALFS, lane), c);
        c = wmma16(xA1, loadB(BIH + (1 * 8 + tile) * FRAG_HALFS, lane), c);
        c = wmma16(hA,  loadB(BHH + tile * FRAG_HALFS, lane), c);
        acc[q] = c;
      }
      lstm_pointwise(acc[0], acc[1], acc[2], acc[3], hc[p], cc[p]);
      sum[p] += hc[p];
    }
    // C-layout h -> A-layout fragment via per-wave LDS bounce
    storeC_lds(wbuf, lane, 0,  hc[0]);
    storeC_lds(wbuf, lane, 16, hc[1]);
    __syncthreads();
    hA = loadA_lds(wbuf, lane, 0);
    __syncthreads();
  }
}

// ---------------- fused kernel ----------------

__global__ void __launch_bounds__(THREADS)
mt_fused(const float* __restrict__ x,
         const float* __restrict__ Wihf, const float* __restrict__ Whhf,
         const float* __restrict__ bihf, const float* __restrict__ bhhf,
         const float* __restrict__ Wihr, const float* __restrict__ Whhr,
         const float* __restrict__ bihr, const float* __restrict__ bhhr,
         const float* __restrict__ Wcih, const float* __restrict__ Wchh,
         const float* __restrict__ bcih, const float* __restrict__ bchh,
         const float* __restrict__ Wo,   const float* __restrict__ bo,
         float* __restrict__ out) {
  extern __shared__ char smem_raw[];
  f16* BIHF = (f16*)smem_raw;                 // 16 frags
  f16* BIHR = BIHF + 16 * FRAG_HALFS;         // 16
  f16* BHHF = BIHR + 16 * FRAG_HALFS;         // 8
  f16* BHHR = BHHF + 8 * FRAG_HALFS;          // 8
  f16* WCIHB = BHHR + 8 * FRAG_HALFS;         // 32
  f16* WCHHB = WCIHB + 32 * FRAG_HALFS;       // 32
  f16* WOB  = WCHHB + 32 * FRAG_HALFS;        // 2
  f16* WBUF = WOB + 2 * FRAG_HALFS;           // 8 waves x 16x64
  float* BIAS = (float*)(WBUF + WBUF_HALFS);
  float* BF = BIAS;            // 128
  float* BR = BF + 128;        // 128
  float* BC = BR + 128;        // 256
  float* BO = BC + 256;        // 16
  float* OBUF = BO + 16;       // 8 waves x 256 f32

  int tid = threadIdx.x;

  // ---- one-time per-block weight packing (f32 -> f16 B-fragments) ----
  packB(Wihf, 4 * NA, VIN, VIN, 8, 2, BIHF, tid, THREADS);
  packB(Wihr, 4 * NA, VIN, VIN, 8, 2, BIHR, tid, THREADS);
  packB(Whhf, 4 * NA, NA, NA, 8, 1, BHHF, tid, THREADS);
  packB(Whhr, 4 * NA, NA, NA, 8, 1, BHHR, tid, THREADS);
  packB(Wcih, 4 * NS, NS, NS, 16, 2, WCIHB, tid, THREADS);
  packB(Wchh, 4 * NS, NS, NS, 16, 2, WCHHB, tid, THREADS);
  packB(Wo, VOUT, NS, NS, 1, 2, WOB, tid, THREADS);
  for (int i = tid; i < 128; i += THREADS) {
    BF[i] = bihf[i] + bhhf[i];
    BR[i] = bihr[i] + bhhr[i];
  }
  for (int i = tid; i < 256; i += THREADS) BC[i] = bcih[i] + bchh[i];
  for (int i = tid; i < 16; i += THREADS) BO[i] = (i < VOUT) ? bo[i] : -1e30f;
  __syncthreads();

  int wave = tid >> 5, lane = tid & 31;
  long rowbase = (long)blockIdx.x * ROWS_PER_BLOCK + (long)wave * ROWS_PER_WAVE;
  f16* wbuf = WBUF + wave * 16 * 64;
  float* obuf = OBUF + wave * 256;

  // ---- BiLSTM encoder; ctx = sum_t a (attention weights are exactly 1) ----
  v8f sumF[2], sumR[2];
  encoder_dir(x, rowbase, lane, false, BIHF, BHHF, BF, wbuf, sumF);
  encoder_dir(x, rowbase, lane, true,  BIHR, BHHR, BR, wbuf, sumR);

  storeC_lds(wbuf, lane, 0,  sumF[0]);
  storeC_lds(wbuf, lane, 16, sumF[1]);
  storeC_lds(wbuf, lane, 32, sumR[0]);
  storeC_lds(wbuf, lane, 48, sumR[1]);
  __syncthreads();
  v16h ctxA0 = loadA_lds(wbuf, lane, 0);
  v16h ctxA1 = loadA_lds(wbuf, lane, 32);
  __syncthreads();

  // ---- decoder: 10 LSTMCell steps + per-step softmax output ----
  v8f sc[4] = {zero8(), zero8(), zero8(), zero8()};
  v8f c2[4] = {zero8(), zero8(), zero8(), zero8()};
  v16h sA0 = zero16h(), sA1 = zero16h();

#pragma unroll 1
  for (int t = 0; t < TY; ++t) {
    // gates[16x256]: tiles {i:0-3 f:4-7 g:8-11 o:12-15}; pass p -> h cols 16p..
#pragma unroll
    for (int p = 0; p < 4; ++p) {
      v8f acc[4];
#pragma unroll
      for (int q = 0; q < 4; ++q) {
        int tile = 4 * q + p;
        v8f c = splat8(BC[tile * 16 + (lane & 15)]);
        c = wmma16(ctxA0, loadB(WCIHB + (0 * 16 + tile) * FRAG_HALFS, lane), c);
        c = wmma16(ctxA1, loadB(WCIHB + (1 * 16 + tile) * FRAG_HALFS, lane), c);
        c = wmma16(sA0,   loadB(WCHHB + (0 * 16 + tile) * FRAG_HALFS, lane), c);
        c = wmma16(sA1,   loadB(WCHHB + (1 * 16 + tile) * FRAG_HALFS, lane), c);
        acc[q] = c;
      }
      lstm_pointwise(acc[0], acc[1], acc[2], acc[3], sc[p], c2[p]);
    }
    // new s -> A fragments via LDS bounce
    storeC_lds(wbuf, lane, 0,  sc[0]);
    storeC_lds(wbuf, lane, 16, sc[1]);
    storeC_lds(wbuf, lane, 32, sc[2]);
    storeC_lds(wbuf, lane, 48, sc[3]);
    __syncthreads();
    sA0 = loadA_lds(wbuf, lane, 0);
    sA1 = loadA_lds(wbuf, lane, 32);
    __syncthreads();

    // logits = s @ Wo^T + bo  (N padded 11->16)
    v8f lg = splat8(BO[lane & 15]);
    lg = wmma16(sA0, loadB(WOB, lane), lg);
    lg = wmma16(sA1, loadB(WOB + FRAG_HALFS, lane), lg);
    {
      int n = lane & 15, hs = lane >> 4;
#pragma unroll
      for (int j = 0; j < 8; ++j) obuf[(j + 8 * hs) * 16 + n] = lg[j];
    }
    __syncthreads();
    if (lane < 16) {
      float v[VOUT];
      float mx = -1e30f;
#pragma unroll
      for (int k = 0; k < VOUT; ++k) {
        v[k] = obuf[lane * 16 + k];
        mx = fmaxf(mx, v[k]);
      }
      float s = 0.0f;
#pragma unroll
      for (int k = 0; k < VOUT; ++k) {
        v[k] = __expf(v[k] - mx);
        s += v[k];
      }
      float inv = 1.0f / s;
      float* o = out + ((rowbase + lane) * TY + t) * VOUT;
#pragma unroll
      for (int k = 0; k < VOUT; ++k) o[k] = v[k] * inv;
    }
    __syncthreads();
  }
}

// ---------------- launch ----------------

extern "C" void kernel_launch(void* const* d_in, const int* in_sizes, int n_in,
                              void* d_out, int out_size, void* d_ws, size_t ws_size,
                              hipStream_t stream) {
  (void)in_sizes; (void)n_in; (void)out_size; (void)d_ws; (void)ws_size;
  const float* x    = (const float*)d_in[0];
  const float* Wihf = (const float*)d_in[1];
  const float* Whhf = (const float*)d_in[2];
  const float* bihf = (const float*)d_in[3];
  const float* bhhf = (const float*)d_in[4];
  const float* Wihr = (const float*)d_in[5];
  const float* Whhr = (const float*)d_in[6];
  const float* bihr = (const float*)d_in[7];
  const float* bhhr = (const float*)d_in[8];
  // d_in[9..12] (Wa1, ba1, Wa2, ba2) are dead: softmax over a size-1 axis == 1,
  // so context = a.sum(axis=1) regardless of the attention MLP.
  const float* Wcih = (const float*)d_in[13];
  const float* Wchh = (const float*)d_in[14];
  const float* bcih = (const float*)d_in[15];
  const float* bchh = (const float*)d_in[16];
  const float* Wo   = (const float*)d_in[17];
  const float* bo   = (const float*)d_in[18];
  float* out = (float*)d_out;

  mt_fused<<<NBLOCKS, THREADS, SMEM_BYTES, stream>>>(
      x, Wihf, Whhf, bihf, bhhf, Wihr, Whhr, bihr, bhhr,
      Wcih, Wchh, bcih, bchh, Wo, bo, out);
}
